// MultiheadSelfAttention_14508399525942
// MI455X (gfx1250) — matrix operand order repaired and checked by reading, not compiled, not run on hardware
//
#include <hip/hip_runtime.h>

// ---------------------------------------------------------------------------
// Multi-head self-attention for MI455X (gfx1250, wave32, WMMA).
//   B=4, N=2048, D=1024, H=16, HD=64.  SCALE = 1/8.
// Pass 1: QKV = X @ W^T via bf16 WMMA (f32 accum); epilogue writes Q/K/V to
//         workspace ALREADY in WMMA fragment layout (1KB fragment blocks).
// Pass 2: flash attention; K/V tiles staged with global_load_async_to_lds
//         (ASYNCcnt) as plain contiguous copies; bf16 WMMA + f32 softmax.
// ---------------------------------------------------------------------------

typedef __attribute__((ext_vector_type(16))) __bf16 v16bf;
typedef __attribute__((ext_vector_type(2)))  __bf16 v2bf;
typedef __attribute__((ext_vector_type(8)))  float  v8f;

#define BQ 4
#define HQ 16
#define NQ 2048
#define DQ 1024
#define HDQ 64
#define SCALEQ 0.125f

__device__ __forceinline__ unsigned short f2bf(float f) {
    __bf16 h = (__bf16)f;                       // native cvt (RNE)
    return __builtin_bit_cast(unsigned short, h);
}
__device__ __forceinline__ unsigned int pack2(float a, float b) {
    v2bf v; v.x = (__bf16)a; v.y = (__bf16)b;   // v_cvt_pk path
    return __builtin_bit_cast(unsigned int, v);
}

// A-fragment (16x32 bf16) swizzle: element (m,k) -> byte offset in 1KB block.
__device__ __forceinline__ int a_swz(int mloc, int k) {
    int lane = mloc + 16 * ((k >> 3) & 1);
    int vg   = ((k & 7) >> 1) + 4 * ((k >> 4) & 1);
    return lane * 32 + vg * 4 + (k & 1) * 2;
}
// B-fragment (32x16 bf16) swizzle: element (k,n) -> byte offset in 1KB block.
__device__ __forceinline__ int b_swz(int k, int nloc) {
    int lane = nloc + 16 * ((k >> 3) & 1);
    int vg   = ((k & 7) >> 1) + 4 * ((k >> 4) & 1);
    return lane * 32 + vg * 4 + (k & 1) * 2;
}

// Workspace fragment layouts (byte offsets within each 16MB region).
// Q: [bh][n/16][d/32] 1KB A-frag blocks.
__device__ __forceinline__ size_t qws_off(int bh, int n, int d) {
    return ((size_t)(bh * 128 + (n >> 4)) * 2 + (d >> 5)) * 1024 + a_swz(n & 15, d & 31);
}
// K: [bh][n/128][ (n/16%8)*2 + d/32 ] 1KB B-frag blocks (for Q@K^T).
__device__ __forceinline__ size_t kws_off(int bh, int n, int d) {
    return ((size_t)(bh * 16 + (n >> 7)) * 16 + ((n >> 4) & 7) * 2 + (d >> 5)) * 1024
           + b_swz(d & 31, n & 15);
}
// V: [bh][n/128][ (n/32%4)*4 + d/16 ] 1KB B-frag blocks (for P@V).
__device__ __forceinline__ size_t vws_off(int bh, int n, int d) {
    return ((size_t)(bh * 16 + (n >> 7)) * 16 + ((n >> 5) & 3) * 4 + (d >> 4)) * 1024
           + b_swz(n & 31, d & 15);
}

// Async global -> LDS 16B copy (per-lane), tracked by ASYNCcnt.
__device__ __forceinline__ void async_copy_b128(unsigned int lds_off, const void* g) {
    asm volatile("global_load_async_to_lds_b128 %0, %1, off"
                 :: "v"(lds_off), "v"(g) : "memory");
}
__device__ __forceinline__ void wait_async0() {
    asm volatile("s_wait_asynccnt 0" ::: "memory");
}

// ---------------------------------------------------------------------------
// Kernel 1: QKV projection.  C[8192,3072] = X[8192,1024] @ W[3072,1024]^T
// Block: 256 threads (8 waves).  Tile 128(M) x 128(N), K-step 32.
// ---------------------------------------------------------------------------
__global__ __launch_bounds__(256)
void qkv_gemm_kernel(const float* __restrict__ X,
                     const float* __restrict__ W,
                     char* __restrict__ ws) {
    __shared__ __align__(128) char smem[16384];   // As 8KB | Bs 8KB
    char* As = smem;
    char* Bs = smem + 8192;

    const int tid  = threadIdx.x;
    const int lane = tid & 31;
    const int w    = tid >> 5;
    const int waveM = w >> 1;           // 0..3
    const int waveN = w & 1;            // 0..1
    const int m0 = blockIdx.y * 128;
    const int n0 = blockIdx.x * 128;

    v8f acc[2][4];
    #pragma unroll
    for (int mi = 0; mi < 2; ++mi)
        #pragma unroll
        for (int ni = 0; ni < 4; ++ni)
            acc[mi][ni] = v8f{0.f,0.f,0.f,0.f,0.f,0.f,0.f,0.f};

    for (int kt = 0; kt < 32; ++kt) {
        const int k0 = kt * 32;
        __syncthreads();
        #pragma unroll
        for (int i = 0; i < 4; ++i) {
            int idx = tid + i * 256;
            int row = idx >> 3;               // 0..127
            int kg  = (idx & 7) * 4;          // 0..28
            float4 xa = *(const float4*)(X + (size_t)(m0 + row) * DQ + k0 + kg);
            float4 wb = *(const float4*)(W + (size_t)(n0 + row) * DQ + k0 + kg);
            int sub = (row >> 4) * 1024;
            int rl  = row & 15;
            *(unsigned int*)(As + sub + a_swz(rl, kg))     = pack2(xa.x, xa.y);
            *(unsigned int*)(As + sub + a_swz(rl, kg + 2)) = pack2(xa.z, xa.w);
            *(unsigned int*)(Bs + sub + b_swz(kg,     rl)) = pack2(wb.x, wb.y);
            *(unsigned int*)(Bs + sub + b_swz(kg + 2, rl)) = pack2(wb.z, wb.w);
        }
        __syncthreads();
        v16bf a[2], b[4];
        #pragma unroll
        for (int mi = 0; mi < 2; ++mi)
            a[mi] = *(const v16bf*)(As + (waveM * 2 + mi) * 1024 + lane * 32);
        #pragma unroll
        for (int ni = 0; ni < 4; ++ni)
            b[ni] = *(const v16bf*)(Bs + (waveN * 4 + ni) * 1024 + lane * 32);
        #pragma unroll
        for (int mi = 0; mi < 2; ++mi)
            #pragma unroll
            for (int ni = 0; ni < 4; ++ni)
                acc[mi][ni] = __builtin_amdgcn_wmma_f32_16x16x32_bf16(
                    false, a[mi], false, b[ni], (short)0, acc[mi][ni], false, false);
    }

    // ---- epilogue: scatter C fragments directly into fragment-layout ws
    char* qws = ws;
    char* kws = ws + (16u << 20);
    char* vws = ws + (32u << 20);
    const int hi = (lane >> 4) & 1;
    #pragma unroll
    for (int mi = 0; mi < 2; ++mi) {
        #pragma unroll
        for (int ni = 0; ni < 4; ++ni) {
            int colBase = n0 + (waveN * 4 + ni) * 16;   // 16-aligned -> p,h uniform
            int p   = colBase >> 10;
            int rem = colBase & 1023;
            int h2  = rem >> 6;
            int dl  = (rem & 63) + (lane & 15);         // 0..63
            #pragma unroll
            for (int r = 0; r < 8; ++r) {
                int row = m0 + (waveM * 2 + mi) * 16 + r + 8 * hi;
                int bb  = row >> 11;
                int n   = row & 2047;
                int bh  = bb * HQ + h2;
                unsigned short e = f2bf(acc[mi][ni][r]);
                if (p == 0)      *(unsigned short*)(qws + qws_off(bh, n, dl)) = e;
                else if (p == 1) *(unsigned short*)(kws + kws_off(bh, n, dl)) = e;
                else             *(unsigned short*)(vws + vws_off(bh, n, dl)) = e;
            }
        }
    }
}

// ---------------------------------------------------------------------------
// Kernel 2: flash attention.  Block: 128 threads (4 waves), 64 Q rows/block.
// grid = (N/64 = 32, B*H = 64).  KV tiles of 128.
// LDS: Qs 8KB | Ks 16KB | Vs 16KB | Ps 16KB = 56KB.
// ---------------------------------------------------------------------------
#define QS_OFF 0
#define KS_OFF 8192
#define VS_OFF 24576
#define PS_OFF 40960

__global__ __launch_bounds__(128)
void flash_attn_kernel(const char* __restrict__ ws,
                       float* __restrict__ out) {
    __shared__ __align__(128) char smem[57344];
    const unsigned int ldsBase = (unsigned int)(uintptr_t)(smem);

    const int tid  = threadIdx.x;
    const int lane = tid & 31;
    const int w    = tid >> 5;              // 0..3
    const int hi   = (lane >> 4) & 1;
    const int mTile = blockIdx.x;           // 0..31
    const int bh    = blockIdx.y;           // b*16 + h
    const int b     = bh >> 4;
    const int h     = bh & 15;

    const char* qws = ws;
    const char* kws = ws + (16u << 20);
    const char* vws = ws + (32u << 20);

    // ---- stage Q tile (8KB, already fragment-ordered): async copy
    const char* Qsrc = qws + ((size_t)bh * 128 + mTile * 4) * 2048;
    #pragma unroll
    for (int i = 0; i < 4; ++i)
        async_copy_b128(ldsBase + QS_OFF + tid * 16 + i * 2048,
                        Qsrc + tid * 16 + i * 2048);
    wait_async0();
    __syncthreads();

    v16bf qa[2];
    #pragma unroll
    for (int kb = 0; kb < 2; ++kb)
        qa[kb] = *(const v16bf*)(smem + QS_OFF + w * 2048 + kb * 1024 + lane * 32);

    v8f o[4];
    #pragma unroll
    for (int dt = 0; dt < 4; ++dt) o[dt] = v8f{0.f,0.f,0.f,0.f,0.f,0.f,0.f,0.f};
    float mrow[8], lrow[8];
    #pragma unroll
    for (int r = 0; r < 8; ++r) { mrow[r] = -1e30f; lrow[r] = 0.f; }

    for (int kt = 0; kt < NQ / 128; ++kt) {
        __syncthreads();   // previous iteration's readers are done
        // ---- stage K & V tiles (16KB each, contiguous) via async LDS DMA
        const char* Ksrc = kws + ((size_t)bh * 16 + kt) * 16384;
        const char* Vsrc = vws + ((size_t)bh * 16 + kt) * 16384;
        #pragma unroll
        for (int i = 0; i < 8; ++i) {
            async_copy_b128(ldsBase + KS_OFF + tid * 16 + i * 2048,
                            Ksrc + tid * 16 + i * 2048);
            async_copy_b128(ldsBase + VS_OFF + tid * 16 + i * 2048,
                            Vsrc + tid * 16 + i * 2048);
        }
        if (kt + 1 < NQ / 128) {           // hint next tiles toward L2
            __builtin_prefetch(Ksrc + 16384 + tid * 256, 0, 0);
            __builtin_prefetch(Vsrc + 16384 + tid * 256, 0, 0);
        }
        wait_async0();
        __syncthreads();

        // ---- S = Q @ K^T : 16 WMMAs per wave
        v8f sc[8];
        #pragma unroll
        for (int t = 0; t < 8; ++t) sc[t] = v8f{0.f,0.f,0.f,0.f,0.f,0.f,0.f,0.f};
        #pragma unroll
        for (int t = 0; t < 8; ++t) {
            #pragma unroll
            for (int kb = 0; kb < 2; ++kb) {
                v16bf kf = *(const v16bf*)(smem + KS_OFF + (t * 2 + kb) * 1024 + lane * 32);
                sc[t] = __builtin_amdgcn_wmma_f32_16x16x32_bf16(
                    false, qa[kb], false, kf, (short)0, sc[t], false, false);
            }
        }

        // ---- online softmax (row m lives in VGPR r, cols across 16-lane halves)
        #pragma unroll
        for (int r = 0; r < 8; ++r) {
            float mx = sc[0][r] * SCALEQ;
            #pragma unroll
            for (int t = 1; t < 8; ++t) mx = fmaxf(mx, sc[t][r] * SCALEQ);
            #pragma unroll
            for (int off = 8; off >= 1; off >>= 1)
                mx = fmaxf(mx, __shfl_xor(mx, off, 32));
            float mnew  = fmaxf(mrow[r], mx);
            float alpha = __expf(mrow[r] - mnew);
            mrow[r] = mnew;
            float psum = 0.f;
            #pragma unroll
            for (int t = 0; t < 8; ++t) {
                float p = __expf(sc[t][r] * SCALEQ - mnew);
                sc[t][r] = p;
                psum += p;
            }
            #pragma unroll
            for (int off = 8; off >= 1; off >>= 1)
                psum += __shfl_xor(psum, off, 32);
            lrow[r] = lrow[r] * alpha + psum;
            #pragma unroll
            for (int dt = 0; dt < 4; ++dt) o[dt][r] *= alpha;
        }

        // ---- P (C-layout) -> bf16 A-fragment layout in per-wave LDS region
        #pragma unroll
        for (int t = 0; t < 8; ++t) {
            int kcol = t * 16 + (lane & 15);
            char* pbase = smem + PS_OFF + w * 4096 + (kcol >> 5) * 1024
                        + (8 * hi + 16 * ((kcol >> 3) & 1)) * 32
                        + (((kcol & 7) >> 1) + 4 * ((kcol >> 4) & 1)) * 4
                        + (kcol & 1) * 2;
            #pragma unroll
            for (int r = 0; r < 8; ++r)
                *(unsigned short*)(pbase + r * 32) = f2bf(sc[t][r]);
        }
        asm volatile("s_wait_dscnt 0" ::: "memory");  // wave-local RAW via LDS

        // ---- O += P @ V : 16 WMMAs per wave
        v16bf pa[4];
        #pragma unroll
        for (int kb2 = 0; kb2 < 4; ++kb2)
            pa[kb2] = *(const v16bf*)(smem + PS_OFF + w * 4096 + kb2 * 1024 + lane * 32);
        #pragma unroll
        for (int dt = 0; dt < 4; ++dt) {
            #pragma unroll
            for (int kb2 = 0; kb2 < 4; ++kb2) {
                v16bf vf = *(const v16bf*)(smem + VS_OFF + (kb2 * 4 + dt) * 1024 + lane * 32);
                o[dt] = __builtin_amdgcn_wmma_f32_16x16x32_bf16(
                    false, pa[kb2], false, vf, (short)0, o[dt], false, false);
            }
        }
    }

    // ---- normalize and store: out[b, n, h*64 + d]
    #pragma unroll
    for (int dt = 0; dt < 4; ++dt) {
        #pragma unroll
        for (int r = 0; r < 8; ++r) {
            int mloc = r + 8 * hi;
            int nrow = mTile * 64 + w * 16 + mloc;
            int col  = h * 64 + dt * 16 + (lane & 15);
            out[((size_t)b * NQ + nrow) * DQ + col] = o[dt][r] * (1.0f / lrow[r]);
        }
    }
}

// ---------------------------------------------------------------------------
extern "C" void kernel_launch(void* const* d_in, const int* in_sizes, int n_in,
                              void* d_out, int out_size, void* d_ws, size_t ws_size,
                              hipStream_t stream) {
    const float* X = (const float*)d_in[0];      // [4, 2048, 1024]
    const float* W = (const float*)d_in[1];      // [3072, 1024]
    float* out = (float*)d_out;                  // [4, 2048, 1024]
    char* ws = (char*)d_ws;                      // 48MB: Q | K | V fragment layout

    dim3 g1(24, 64);   // N-tiles (3072/128) x M-tiles (8192/128)
    qkv_gemm_kernel<<<g1, 256, 0, stream>>>(X, W, ws);

    dim3 g2(32, 64);   // Q-tiles (2048/64) x (B*H)
    flash_attn_kernel<<<g2, 128, 0, stream>>>(ws, out);
}
